// Decoder_46256797778386
// MI455X (gfx1250) — matrix operand order repaired and checked by reading, not compiled, not run on hardware
//
#include <hip/hip_runtime.h>
#include <hip/hip_bf16.h>
#include <math.h>

// ---------------------------------------------------------------------------
// Transformer decoder forward (4 layers) for MI455X / gfx1250, wave32 + WMMA.
// All GEMMs use V_WMMA_F32_16X16X32_BF16 (bf16 A/B, f32 accumulate).
// ---------------------------------------------------------------------------

typedef __bf16 bf16;
typedef __attribute__((ext_vector_type(16))) __bf16 v16bf;
typedef __attribute__((ext_vector_type(8)))  __bf16 v8bf;
typedef __attribute__((ext_vector_type(2)))  __bf16 v2bf;
typedef __attribute__((ext_vector_type(8)))  float  v8f;

#define E_  1024
#define S_  1024
#define NB  4
#define HH  16
#define DD  64
#define FF_ 4096
#define VV  32000
#define ROWS 4096   // NB * S_

// ds_swizzle_b32 group-of-32 XOR butterfly: offset = {xor[14:10], or[9:5]=0, and[4:0]=0x1F}
#define SWZ_XOR(x, m) \
  __int_as_float(__builtin_amdgcn_ds_swizzle(__float_as_int(x), ((m) << 10) | 0x1F))

__device__ __forceinline__ v16bf join8(v8bf lo, v8bf hi) {
  return __builtin_shufflevector(lo, hi, 0,1,2,3,4,5,6,7,8,9,10,11,12,13,14,15);
}
__device__ __forceinline__ v8f vzero8() {
  v8f z = {0.f,0.f,0.f,0.f,0.f,0.f,0.f,0.f};
  return z;
}
// pack two f32 -> 2xbf16, single 32-bit LDS store (v_cvt_pk_bf16_f32 capable)
__device__ __forceinline__ void st2(__bf16* p, float a, float b) {
  v2bf v; v[0] = (bf16)a; v[1] = (bf16)b;
  *(v2bf*)p = v;
}

// ---------------------------------------------------------------------------
// Embedding + interleaved sin/cos positional encoding. One wave per row.
// ---------------------------------------------------------------------------
__global__ __launch_bounds__(256) void embed_pos_kernel(
    const int* __restrict__ xk, const float* __restrict__ emb,
    float* __restrict__ out) {
  int wid = threadIdx.x >> 5, lane = threadIdx.x & 31;
  int row = blockIdx.x * 8 + wid;           // 0..4095
  int s = row & (S_ - 1);
  int tok = xk[row];
  const float* er = emb + (size_t)tok * E_;
  float* orow = out + (size_t)row * E_;
#pragma unroll
  for (int j = 0; j < 32; ++j) {
    int e = j * 32 + lane;
    float ang = (float)s * __powf(10000.0f, -(float)(e & ~1) * (1.0f / (float)E_));
    float pe = (e & 1) ? __cosf(ang) : __sinf(ang);
    orow[e] = er[e] + pe;
  }
}

// ---------------------------------------------------------------------------
// Tile staging: f32 global -> bf16 LDS (A row-major [m][k], B transposed
// [n][k]); rows padded to 40 bf16 (80 B) for conflict-free ds_load_b128.
// Conversions packed in pairs -> 32-bit LDS stores.
// ---------------------------------------------------------------------------
__device__ __forceinline__ void stage_tiles(
    const float* __restrict__ A, const float* __restrict__ B,
    __bf16* Als, __bf16* Bls, int bm, int bn, int K, int Nc, int k0, int tid) {
  // ---- A tile: 128x32; pairs packed along k ------------------------------
#pragma unroll
  for (int j = 0; j < 4; ++j) {
    int idx = j * 256 + tid;
    int r = idx >> 3, c4 = (idx & 7) * 4;
    float4 v = *(const float4*)(A + (size_t)(bm + r) * K + k0 + c4);
    st2(&Als[r * 40 + c4],     v.x, v.y);
    st2(&Als[r * 40 + c4 + 2], v.z, v.w);
  }
  // ---- B tile: 32x128 -> [n][k]; two k-rows per thread, packed along k ---
#pragma unroll
  for (int j = 0; j < 2; ++j) {
    int idx = j * 256 + tid;             // 0..511
    int n4 = (idx & 31) * 4;             // n in {0,4,...,124}
    int k  = (idx >> 5) * 2;             // even k in [0,32)
    const float* b0 = B + (size_t)(k0 + k) * Nc + bn + n4;
    float4 r0 = *(const float4*)b0;
    float4 r1 = *(const float4*)(b0 + Nc);
    st2(&Bls[(n4 + 0) * 40 + k], r0.x, r1.x);
    st2(&Bls[(n4 + 1) * 40 + k], r0.y, r1.y);
    st2(&Bls[(n4 + 2) * 40 + k], r0.z, r1.z);
    st2(&Bls[(n4 + 3) * 40 + k], r0.w, r1.w);
  }
}

// ---------------------------------------------------------------------------
// Tiled bf16 WMMA GEMM:  C[4096,Nc] = A[4096,K] * B[K,Nc] + bias (+epilogue)
// MODE: 0 = bias only, 1 = bias + residual, 2 = bias + ReLU.
// Block = 256 threads (8 waves), tile 128x128, K-step 32, double-buffered LDS
// so the next tile's global loads overlap the current tile's WMMAs; one
// barrier per K-step.  Wave (wm,wn) computes 32x64 = 2x4 WMMA 16x16 tiles.
// ---------------------------------------------------------------------------
template <int MODE>
__global__ __launch_bounds__(256) void gemm_bf16_kernel(
    const float* __restrict__ A, const float* __restrict__ B,
    const float* __restrict__ bias, const float* __restrict__ resid,
    float* __restrict__ C, int K, int Nc) {
  __shared__ __bf16 Als[2][128 * 40];
  __shared__ __bf16 Bls[2][128 * 40];
  int tid  = threadIdx.x;
  int lane = tid & 31, wid = tid >> 5;
  int half = lane >> 4, lr = lane & 15;
  int wm = wid & 3, wn = wid >> 2;
  int bm = blockIdx.y * 128, bn = blockIdx.x * 128;

  v8f acc[2][4];
#pragma unroll
  for (int t = 0; t < 2; ++t)
#pragma unroll
    for (int u = 0; u < 4; ++u) acc[t][u] = vzero8();

  stage_tiles(A, B, Als[0], Bls[0], bm, bn, K, Nc, 0, tid);

  int nk = K >> 5;
  for (int kt = 0; kt < nk; ++kt) {
    __syncthreads();                       // staged[buf] visible; buf^1 free
    int buf = kt & 1;
    if (kt + 1 < nk)
      stage_tiles(A, B, Als[buf ^ 1], Bls[buf ^ 1], bm, bn, K, Nc,
                  (kt + 1) << 5, tid);
    // ---- fragments (CDNA5 16-bit A/B VGPR layouts) ------------------------
    v16bf af[2], bfv[4];
#pragma unroll
    for (int t = 0; t < 2; ++t) {
      int base = (wm * 32 + t * 16 + lr) * 40;
      v8bf lo = *(const v8bf*)&Als[buf][base + 8 * half];      // K=8h+0..7
      v8bf hi = *(const v8bf*)&Als[buf][base + 16 + 8 * half]; // K=16+8h+0..7
      af[t] = join8(lo, hi);
    }
#pragma unroll
    for (int u = 0; u < 4; ++u) {
      int base = (wn * 64 + u * 16 + lr) * 40;
      v8bf lo = *(const v8bf*)&Bls[buf][base + 8 * half];      // K=8h+0..7
      v8bf hi = *(const v8bf*)&Bls[buf][base + 16 + 8 * half]; // K=16+8h+0..7
      bfv[u] = join8(lo, hi);
    }
#pragma unroll
    for (int t = 0; t < 2; ++t)
#pragma unroll
      for (int u = 0; u < 4; ++u)
        acc[t][u] = __builtin_amdgcn_wmma_f32_16x16x32_bf16(
            false, af[t], false, bfv[u], (short)0, acc[t][u], false, false);
  }
  // ---- epilogue: lane holds col n=lr, rows r + 8*half; pointer-stepped ----
#pragma unroll
  for (int t = 0; t < 2; ++t) {
#pragma unroll
    for (int u = 0; u < 4; ++u) {
      int col  = bn + wn * 64 + u * 16 + lr;
      int row0 = bm + wm * 32 + t * 16 + 8 * half;
      float bi = bias[col];
      float* cp = C + (size_t)row0 * Nc + col;
      const float* rp = (MODE == 1) ? resid + (size_t)row0 * Nc + col : nullptr;
#pragma unroll
      for (int r = 0; r < 8; ++r) {
        float v = acc[t][u][r] + bi;
        if (MODE == 1) { v += *rp; rp += Nc; }
        if (MODE == 2) v = fmaxf(v, 0.0f);
        *cp = v;
        cp += Nc;
      }
    }
  }
}

// ---------------------------------------------------------------------------
// Flash-style causal attention. Grid (qblock=8, head=16, batch=4), 256 thr.
// Each wave owns 16 q rows; loop over 32-key blocks with running max/sum.
// S = Q*K^T and O += P*V via bf16 WMMA; softmax math in f32.
// All fragments preloaded into distinct registers before the WMMA chains so
// the scheduler can interleave s_wait_dscnt with independent WMMAs; the two
// S accumulator chains alternate to break D->C serialization.
// Reference scale = 1/sqrt(E) = 1/32.
// ---------------------------------------------------------------------------
__global__ __launch_bounds__(256) void attn_kernel(
    const float* __restrict__ Q, const float* __restrict__ Kg,
    const float* __restrict__ Vg, float* __restrict__ O) {
  __shared__ __bf16 Kl[32 * 72];        // [key][d]  (K-contig per B column)
  __shared__ __bf16 Vl[64 * 40];        // [d][key]  (transposed)
  __shared__ __bf16 Pl[8 * 16 * 40];    // per-wave P transpose buffer
  int tid  = threadIdx.x;
  int lane = tid & 31;
  int wid_s = __builtin_amdgcn_readfirstlane(tid) >> 5;   // scalar wave id
  int half = lane >> 4, lr = lane & 15;
  int qb = blockIdx.x, h = blockIdx.y, n = blockIdx.z;
  int qrow0   = qb * 128 + wid_s * 16;                    // scalar
  int rowbase = n * S_;
  const float scale = 0.03125f;         // 1/sqrt(1024)

  // Q fragments: 16 rows x 64 d -> two 16x32 A-frags (lane row = lr)
  v16bf qf[2];
  {
    const float* qr = Q + (size_t)(rowbase + qrow0 + lr) * E_ + h * DD;
#pragma unroll
    for (int c = 0; c < 2; ++c) {
      v16bf v;
#pragma unroll
      for (int i = 0; i < 8; ++i) v[i]     = (bf16)qr[c * 32 + 8 * half + i];
#pragma unroll
      for (int i = 0; i < 8; ++i) v[8 + i] = (bf16)qr[c * 32 + 16 + 8 * half + i];
      qf[c] = v;
    }
  }

  v8f oacc[4];
#pragma unroll
  for (int u = 0; u < 4; ++u) oacc[u] = vzero8();
  float mrun[8], lrun[8];
#pragma unroll
  for (int r = 0; r < 8; ++r) { mrun[r] = -1e30f; lrun[r] = 0.0f; }

  int kend = qb * 128 + 128;
  for (int kb = 0; kb < kend; kb += 32) {
    __syncthreads();
    // ---- stage K [key][d] packed along d; V [d][key] packed along key ----
#pragma unroll
    for (int j = 0; j < 4; ++j) {
      int idx = j * 256 + tid;            // 0..1023
      int d2  = (idx & 31) * 2;           // even d
      int key = idx >> 5;                 // 0..31
      const float* kp = Kg + (size_t)(rowbase + kb + key) * E_ + h * DD + d2;
      st2(&Kl[key * 72 + d2], kp[0], kp[1]);
    }
#pragma unroll
    for (int j = 0; j < 4; ++j) {
      int idx = j * 256 + tid;            // 0..1023
      int d   = idx & 63;
      int key = (idx >> 6) * 2;           // even key
      const float* vp = Vg + (size_t)(rowbase + kb + key) * E_ + h * DD + d;
      st2(&Vl[d * 40 + key], vp[0], vp[E_]);
    }
    __syncthreads();
    if (kb <= qrow0 + 15) {               // scalar guard: EXEC stays all-ones
      // ---- preload all 4 K fragments, then 4 WMMAs on alternating chains --
      v16bf kf[2][2];
#pragma unroll
      for (int kt = 0; kt < 2; ++kt)
#pragma unroll
        for (int c = 0; c < 2; ++c) {
          int base = (kt * 16 + lr) * 72 + c * 32 + 8 * half;
          kf[kt][c] = join8(*(const v8bf*)&Kl[base],
                            *(const v8bf*)&Kl[base + 16]);
        }
      v8f s0 = __builtin_amdgcn_wmma_f32_16x16x32_bf16(
          false, qf[0], false, kf[0][0], (short)0, vzero8(), false, false);
      v8f s1 = __builtin_amdgcn_wmma_f32_16x16x32_bf16(
          false, qf[0], false, kf[1][0], (short)0, vzero8(), false, false);
      s0 = __builtin_amdgcn_wmma_f32_16x16x32_bf16(
          false, qf[1], false, kf[0][1], (short)0, s0, false, false);
      s1 = __builtin_amdgcn_wmma_f32_16x16x32_bf16(
          false, qf[1], false, kf[1][1], (short)0, s1, false, false);
      // ---- online softmax update (rows r+8*half, cols lr within half) -----
      float e0[8], e1[8];
#pragma unroll
      for (int r = 0; r < 8; ++r) {
        int qa = qrow0 + r + 8 * half;
        float v0 = (kb + lr      <= qa) ? s0[r] * scale : -1e30f;
        float v1 = (kb + 16 + lr <= qa) ? s1[r] * scale : -1e30f;
        float m = fmaxf(v0, v1);
        m = fmaxf(m, SWZ_XOR(m, 1));
        m = fmaxf(m, SWZ_XOR(m, 2));
        m = fmaxf(m, SWZ_XOR(m, 4));
        m = fmaxf(m, SWZ_XOR(m, 8));
        float mnew = fmaxf(mrun[r], m);
        float corr = __expf(mrun[r] - mnew);
        float p0 = __expf(v0 - mnew);
        float p1 = __expf(v1 - mnew);
        float sum = p0 + p1;
        sum += SWZ_XOR(sum, 1);
        sum += SWZ_XOR(sum, 2);
        sum += SWZ_XOR(sum, 4);
        sum += SWZ_XOR(sum, 8);
        lrun[r] = lrun[r] * corr + sum;
        mrun[r] = mnew;
#pragma unroll
        for (int u = 0; u < 4; ++u) oacc[u][r] *= corr;
        e0[r] = p0; e1[r] = p1;
      }
      // ---- transpose P through per-wave LDS -> A-fragment for P*V ---------
      int pbase = wid_s * 16 * 40;
#pragma unroll
      for (int r = 0; r < 8; ++r) {
        int m = r + 8 * half;
        Pl[pbase + m * 40 + lr]      = (bf16)e0[r];
        Pl[pbase + m * 40 + 16 + lr] = (bf16)e1[r];
      }
      asm volatile("s_wait_dscnt 0" ::: "memory");   // same-wave ds RAW guard
      v16bf pf = join8(*(const v8bf*)&Pl[pbase + lr * 40 + 8 * half],
                       *(const v8bf*)&Pl[pbase + lr * 40 + 16 + 8 * half]);
      // ---- preload all 4 V fragments, then 4 independent WMMAs ------------
      v16bf vf[4];
#pragma unroll
      for (int u = 0; u < 4; ++u) {
        int base = (u * 16 + lr) * 40 + 8 * half;
        vf[u] = join8(*(const v8bf*)&Vl[base], *(const v8bf*)&Vl[base + 16]);
      }
#pragma unroll
      for (int u = 0; u < 4; ++u)
        oacc[u] = __builtin_amdgcn_wmma_f32_16x16x32_bf16(
            false, pf, false, vf[u], (short)0, oacc[u], false, false);
    }
  }
  // O = acc * (1/l), written to [n,s, h*64+d]
  float inv[8];
#pragma unroll
  for (int r = 0; r < 8; ++r) inv[r] = __builtin_amdgcn_rcpf(lrun[r]);
#pragma unroll
  for (int u = 0; u < 4; ++u) {
    int col = h * DD + u * 16 + lr;
    float* op = O + (size_t)(rowbase + qrow0 + 8 * half) * E_ + col;
#pragma unroll
    for (int r = 0; r < 8; ++r) {
      *op = oacc[u][r] * inv[r];
      op += E_;
    }
  }
}

// ---------------------------------------------------------------------------
// LayerNorm: one wave per 1024-wide row, ds_swizzle butterfly reductions.
// ---------------------------------------------------------------------------
__global__ __launch_bounds__(256) void ln_kernel(
    const float* __restrict__ X, const float* __restrict__ g,
    const float* __restrict__ b, float* __restrict__ Y) {
  int wid = threadIdx.x >> 5, lane = threadIdx.x & 31;
  int row = blockIdx.x * 8 + wid;
  const float* xr = X + (size_t)row * E_;
  float x[32];
  float sum = 0.f;
#pragma unroll
  for (int j = 0; j < 32; ++j) { x[j] = xr[j * 32 + lane]; sum += x[j]; }
  sum += SWZ_XOR(sum, 1);
  sum += SWZ_XOR(sum, 2);
  sum += SWZ_XOR(sum, 4);
  sum += SWZ_XOR(sum, 8);
  sum += SWZ_XOR(sum, 16);
  float mean = sum * (1.0f / 1024.0f);
  float vs = 0.f;
#pragma unroll
  for (int j = 0; j < 32; ++j) { float d = x[j] - mean; vs += d * d; }
  vs += SWZ_XOR(vs, 1);
  vs += SWZ_XOR(vs, 2);
  vs += SWZ_XOR(vs, 4);
  vs += SWZ_XOR(vs, 8);
  vs += SWZ_XOR(vs, 16);
  float rstd = rsqrtf(vs * (1.0f / 1024.0f) + 1e-5f);
  float* yr = Y + (size_t)row * E_;
#pragma unroll
  for (int j = 0; j < 32; ++j) {
    int e = j * 32 + lane;
    yr[e] = (x[j] - mean) * rstd * g[e] + b[e];
  }
}

// ---------------------------------------------------------------------------
extern "C" void kernel_launch(void* const* d_in, const int* in_sizes, int n_in,
                              void* d_out, int out_size, void* d_ws, size_t ws_size,
                              hipStream_t stream) {
  (void)in_sizes; (void)n_in; (void)out_size; (void)ws_size;
  const int*   x     = (const int*)d_in[0];
  const float* emb   = (const float*)d_in[1];
  const float* Wq    = (const float*)d_in[2];
  const float* bq    = (const float*)d_in[3];
  const float* Wk    = (const float*)d_in[4];
  const float* bk    = (const float*)d_in[5];
  const float* Wv    = (const float*)d_in[6];
  const float* bv    = (const float*)d_in[7];
  const float* Wo    = (const float*)d_in[8];
  const float* bo    = (const float*)d_in[9];
  const float* ln1g  = (const float*)d_in[10];
  const float* ln1b  = (const float*)d_in[11];
  const float* ln2g  = (const float*)d_in[12];
  const float* ln2b  = (const float*)d_in[13];
  const float* W1    = (const float*)d_in[14];
  const float* b1    = (const float*)d_in[15];
  const float* W2    = (const float*)d_in[16];
  const float* b2    = (const float*)d_in[17];
  const float* Wout  = (const float*)d_in[18];
  const float* bout  = (const float*)d_in[19];

  float* ws  = (float*)d_ws;
  const size_t RE = (size_t)ROWS * E_;
  float* cur = ws;                 // [4096,1024] residual stream
  float* qb  = cur + RE;
  float* kb2 = qb  + RE;
  float* vb  = kb2 + RE;
  float* att = vb  + RE;
  float* t1  = att + RE;
  float* hb  = t1  + RE;
  float* ffb = hb  + RE;           // [4096,4096]
  float* t2  = ffb + (size_t)ROWS * FF_;

  const dim3 blk(256);
  const dim3 gE(E_ / 128, ROWS / 128);     // (8, 32)
  const dim3 gF(FF_ / 128, ROWS / 128);    // (32, 32)
  const dim3 gV(VV / 128, ROWS / 128);     // (250, 32)

  embed_pos_kernel<<<ROWS / 8, blk, 0, stream>>>(x, emb, cur);

  for (int i = 0; i < 4; ++i) {
    const float* Wqi = Wq + (size_t)i * E_ * E_;
    const float* Wki = Wk + (size_t)i * E_ * E_;
    const float* Wvi = Wv + (size_t)i * E_ * E_;
    const float* Woi = Wo + (size_t)i * E_ * E_;
    const float* W1i = W1 + (size_t)i * E_ * FF_;
    const float* W2i = W2 + (size_t)i * FF_ * E_;

    gemm_bf16_kernel<0><<<gE, blk, 0, stream>>>(cur, Wqi, bq + i * E_, nullptr, qb,  E_, E_);
    gemm_bf16_kernel<0><<<gE, blk, 0, stream>>>(cur, Wki, bk + i * E_, nullptr, kb2, E_, E_);
    gemm_bf16_kernel<0><<<gE, blk, 0, stream>>>(cur, Wvi, bv + i * E_, nullptr, vb,  E_, E_);

    attn_kernel<<<dim3(S_ / 128, HH, NB), blk, 0, stream>>>(qb, kb2, vb, att);

    gemm_bf16_kernel<1><<<gE, blk, 0, stream>>>(att, Woi, bo + i * E_, cur, t1, E_, E_);
    ln_kernel<<<ROWS / 8, blk, 0, stream>>>(t1, ln1g + i * E_, ln1b + i * E_, hb);

    gemm_bf16_kernel<2><<<gF, blk, 0, stream>>>(hb,  W1i, b1 + i * FF_, nullptr, ffb, E_,  FF_);
    gemm_bf16_kernel<1><<<gE, blk, 0, stream>>>(ffb, W2i, b2 + i * E_,  hb,      t2,  FF_, E_);
    ln_kernel<<<ROWS / 8, blk, 0, stream>>>(t2, ln2g + i * E_, ln2b + i * E_, cur);
  }

  gemm_bf16_kernel<0><<<gV, blk, 0, stream>>>(cur, Wout, bout, nullptr, (float*)d_out,
                                              E_, VV);
}
